// LatticeELMoLM_31121333027064
// MI455X (gfx1250) — matrix operand order repaired and checked by reading, not compiled, not run on hardware
//
#include <hip/hip_runtime.h>
#include <hip/hip_bf16.h>

// ---------------------------------------------------------------------------
// LatticeELMoLM for MI455X (gfx1250, wave32, WMMA)
//   B=32, T=128, P=4, D_IN=512, H=512, C=4096, 4C=16384
// Strategy:
//   * bf16 WMMA (v_wmma_f32_16x16x32_bf16) with fp32 accumulate for all GEMMs
//   * A and B operands pre-packed into hardware fragment order so GEMM inner
//     loops are pure b128 loads + WMMA (no inner-loop cvt)
//   * proj_in hoisted into one big GEMM; 2 small kernels per recurrent step;
//     weights (42 MB packed) stay resident in the 192 MB L2 across 128 steps
//   * per-step kernels restructured for minimum serial critical path:
//     gates split across waves (32-WMMA chains, 1024 waves) and split-K with
//     LDS reduction in the projection (32-WMMA chains, 512 waves)
// ---------------------------------------------------------------------------

typedef __bf16 v16bf __attribute__((ext_vector_type(16)));
typedef float  v8f   __attribute__((ext_vector_type(8)));

constexpr int kB  = 32;
constexpr int kT  = 128;
constexpr int kP  = 4;
constexpr int kD  = 512;    // D_IN
constexpr int kH  = 512;
constexpr int kC  = 4096;
constexpr int kG  = 16384;  // 4C
constexpr float kClip = 3.0f;

__device__ __forceinline__ v8f wmma_bf16(v16bf a, v16bf b, v8f c) {
  // D = A(16x32 bf16) * B(32x16 bf16) + C(16x16 f32)
  return __builtin_amdgcn_wmma_f32_16x16x32_bf16(
      /*neg_a=*/false, a, /*neg_b=*/false, b,
      /*c_mod=*/(short)0, c, /*reuse_a=*/false, /*reuse_b=*/false);
}

__device__ __forceinline__ float sigmoidf_(float x) {
  return 1.0f / (1.0f + __expf(-x));
}
__device__ __forceinline__ float clip3(float x) {
  return fminf(fmaxf(x, -kClip), kClip);
}

// Fragment layouts (per CDNA5 ISA 16-bit WMMA tables):
//   A (16x32, MxK): lane%16 = row M; half = lane/16;
//       slot s -> k_in_tile = (s/8)*16 + half*8 + (s%8)
//   B (32x16, KxN): lane%16 = col N; half = lane/16;
//       slot s -> k_in_tile = (s/8)*16 + half*8 + (s%8)
//   C/D (16x16 f32): VGPR r -> M = r + half*8 ; N = lane%16
// Packed buffers store fragments tile-major:
//   index = ((tile_row * Ktiles + kt) * 32 + lane) * 16 + s

// ---------------------------------------------------------------------------
// Pack row-major fp32 W[Ntot, Ktot] into bf16 B-fragments.
// ---------------------------------------------------------------------------
__global__ __launch_bounds__(256)
void pack_w(const float* __restrict__ W, __bf16* __restrict__ out,
            int Ntiles, int Ktiles, int Ktot) {
  const long total = (long)Ntiles * Ktiles * 512;
  const long i = (long)blockIdx.x * 256 + threadIdx.x;
  if (i >= total) return;
  const int  s    = (int)(i & 15);
  const int  lane = (int)((i >> 4) & 31);
  const long tile = i >> 9;
  const int  kt   = (int)(tile % Ktiles);
  const int  nt   = (int)(tile / Ktiles);
  const int  n = nt * 16 + (lane & 15);
  const int  k = kt * 32 + ((s >> 3) << 4) + ((lane >> 4) << 3) + (s & 7);
  out[i] = (__bf16)W[(long)n * Ktot + k];
}

// ---------------------------------------------------------------------------
// Pack row-major fp32 X[Mtot, Ktot] into bf16 A-fragments.
// ---------------------------------------------------------------------------
__global__ __launch_bounds__(256)
void pack_a(const float* __restrict__ X, __bf16* __restrict__ out,
            int Mtiles, int Ktiles, int Ktot) {
  const long total = (long)Mtiles * Ktiles * 512;
  const long i = (long)blockIdx.x * 256 + threadIdx.x;
  if (i >= total) return;
  const int  s    = (int)(i & 15);
  const int  lane = (int)((i >> 4) & 31);
  const long tile = i >> 9;
  const int  kt   = (int)(tile % Ktiles);
  const int  mt   = (int)(tile / Ktiles);
  const int  half = lane >> 4;
  const int  m = mt * 16 + (lane & 15);
  const int  k = kt * 32 + ((s >> 3) << 4) + half * 8 + (s & 7);
  out[i] = (__bf16)X[(long)m * Ktot + k];
}

// ---------------------------------------------------------------------------
// proj_in = x @ W_in^T : [4096, 512] x [512, 16384] -> [B*T, 4C]
// grid = (Nt/8 = 128, Mpairs = 128), block = 256 (8 waves).
// Inner loop: 3 x v16bf loads (b128 pairs) + 2 WMMAs per kt.
// ---------------------------------------------------------------------------
__global__ __launch_bounds__(256)
void gemm_proj_in(const __bf16* __restrict__ pAx,   // x packed A-frags [256,16]
                  const __bf16* __restrict__ pWin,  // B-frags [1024,16]
                  float* __restrict__ proj) {
  const int tid  = threadIdx.x;
  const int wave = tid >> 5;
  const int lane = tid & 31;
  const int half = lane >> 4;
  const int nt = blockIdx.x * 8 + wave;  // 0..1023
  const int mp = blockIdx.y;             // 0..127 (pair of 16-row M tiles)

  const __bf16* a0base = pAx + ((long)(mp * 2 + 0) * 16) * 512 + (lane << 4);
  const __bf16* a1base = pAx + ((long)(mp * 2 + 1) * 16) * 512 + (lane << 4);
  const __bf16* bbase  = pWin + ((long)nt * 16) * 512 + (lane << 4);

  v8f acc0 = {}, acc1 = {};
#pragma unroll 4
  for (int kt = 0; kt < 16; ++kt) {
    if (kt + 1 < 16) __builtin_prefetch(bbase + (kt + 1) * 512, 0, 3);
    const v16bf A0 = *reinterpret_cast<const v16bf*>(a0base + kt * 512);
    const v16bf A1 = *reinterpret_cast<const v16bf*>(a1base + kt * 512);
    const v16bf Bf = *reinterpret_cast<const v16bf*>(bbase  + kt * 512);
    acc0 = wmma_bf16(A0, Bf, acc0);
    acc1 = wmma_bf16(A1, Bf, acc1);
  }
  const int n = nt * 16 + (lane & 15);
#pragma unroll
  for (int r = 0; r < 8; ++r) {
    const int m0 = mp * 32 + half * 8 + r;
    const int m1 = m0 + 16;
    proj[(long)m0 * kG + n] = acc0[r];
    proj[(long)m1 * kG + n] = acc1[r];
  }
}

// ---------------------------------------------------------------------------
// Recurrent step, part 1: gates + memory update.
//   g = proj_in[:,t] + prev_st @ W_state^T + b_state          [32, 16384]
//   mem = clip(sig(g_i)*tanh(g_m) + sig(g_f)*prev_mem)
//   a   = sig(g_o) * tanh(mem)  -> written in A-fragment order for part 2
// Block = 512 threads (16 waves) handles 4 channel tiles; each wave owns one
// (channel tile, gate) pair -> serial chain is only 2 x 16 = 32 WMMAs.
// Gate results meet in LDS for the fused pointwise update.
// grid = 64 blocks x 16 waves = 1024 waves (one per SIMD32 device-wide).
// ---------------------------------------------------------------------------
__global__ __launch_bounds__(512)
void step_gate(const float* __restrict__ proj,
               const int*   __restrict__ prev_idx,
               const float* __restrict__ prev_w,
               const __bf16* __restrict__ pWst,
               const float* __restrict__ b_state,
               const float* __restrict__ st_f32,   // [T,B,H]
               float*       __restrict__ mem_f32,  // [T,B,C] (read <t, write t)
               __bf16*      __restrict__ a_frag,   // [2 Mtiles][256 kt][512]
               int t) {
  __shared__ __bf16 s_afrag[2 * 16 * 512];   // 32 KB: prev_st as A-fragments
  __shared__ float  s_g[16 * 512];           // 32 KB: per-wave gate outputs
  __shared__ int    s_idx[kB * kP];
  __shared__ float  s_w[kB * kP];

  const int tid = threadIdx.x;
  if (tid < kB * kP) {
    const int b = tid >> 2, p = tid & 3;
    s_idx[tid] = prev_idx[(b * kT + t) * kP + p];
    s_w[tid]   = prev_w [(b * kT + t) * kP + p];
  }
  __syncthreads();

  // Cooperative weighted gather of prev_st, stored at its fragment position.
  for (int i = tid; i < kB * kH; i += 512) {
    const int b = i >> 9;     // /512
    const int k = i & 511;
    float acc = 0.0f;
#pragma unroll
    for (int p = 0; p < kP; ++p)
      acc += s_w[b * 4 + p] * st_f32[((long)s_idx[b * 4 + p] * kB + b) * kH + k];
    // (b,k) -> A-fragment position
    const int mt   = b >> 4;
    const int kt   = k >> 5;
    const int kk   = k & 31;
    const int half = (kk >> 3) & 1;
    const int s    = ((kk >> 4) << 3) + (kk & 7);
    const int lane = half * 16 + (b & 15);
    s_afrag[((mt * 16 + kt) * 32 + lane) * 16 + s] = (__bf16)acc;
  }
  __syncthreads();

  const int wave = tid >> 5;
  const int lane = tid & 31;
  const int half = lane >> 4;
  const int ctL  = wave >> 2;                // local channel tile 0..3
  const int g    = wave & 3;                 // gate 0..3
  const int ct   = blockIdx.x * 4 + ctL;     // channel tile 0..255
  const int nt   = g * 256 + ct;             // W_state column tile 0..1023

  const __bf16* bbase = pWst + ((long)nt * 16) * 512 + (lane << 4);
  v8f acc0 = {}, acc1 = {};
#pragma unroll 4
  for (int kt = 0; kt < 16; ++kt) {
    if (kt + 1 < 16) __builtin_prefetch(bbase + (kt + 1) * 512, 0, 3);
    const v16bf A0 = *reinterpret_cast<const v16bf*>(
        &s_afrag[((0 * 16 + kt) * 32 + lane) * 16]);
    const v16bf A1 = *reinterpret_cast<const v16bf*>(
        &s_afrag[((1 * 16 + kt) * 32 + lane) * 16]);
    const v16bf Bf = *reinterpret_cast<const v16bf*>(bbase + kt * 512);
    acc0 = wmma_bf16(A0, Bf, acc0);
    acc1 = wmma_bf16(A1, Bf, acc1);
  }
  // Publish gate tile: s_g[wave*512 + m*16 + n_local]
#pragma unroll
  for (int r = 0; r < 8; ++r) {
    const int m0 = half * 8 + r;
    s_g[wave * 512 + m0 * 16 + (lane & 15)]        = acc0[r];
    s_g[wave * 512 + (m0 + 16) * 16 + (lane & 15)] = acc1[r];
  }
  __syncthreads();

  // Fused pointwise: 4 tiles x 32 rows x 16 cols = 2048 values, 4 per thread.
  for (int j = tid; j < 4 * 512; j += 512) {
    const int ctL2 = j >> 9;
    const int rem  = j & 511;       // m*16 + nl
    const int b    = rem >> 4;
    const int nl   = rem & 15;
    const int wbase = ctL2 * 4;
    const int n  = (blockIdx.x * 4 + ctL2) * 16 + nl;  // channel c
    const long rowoff = ((long)b * kT + t) * kG;

    const float gi = s_g[(wbase + 0) * 512 + rem] + proj[rowoff + 0 * kC + n] + b_state[0 * kC + n];
    const float gf = s_g[(wbase + 1) * 512 + rem] + proj[rowoff + 1 * kC + n] + b_state[1 * kC + n];
    const float gm = s_g[(wbase + 2) * 512 + rem] + proj[rowoff + 2 * kC + n] + b_state[2 * kC + n];
    const float go = s_g[(wbase + 3) * 512 + rem] + proj[rowoff + 3 * kC + n] + b_state[3 * kC + n];
    const float ig = sigmoidf_(gi);
    const float fg = sigmoidf_(gf);
    const float mi = tanhf(gm);
    const float og = sigmoidf_(go);

    float pm = 0.0f;
#pragma unroll
    for (int p = 0; p < kP; ++p)
      pm += s_w[b * 4 + p] * mem_f32[((long)s_idx[b * 4 + p] * kB + b) * kC + n];

    const float mem = clip3(ig * mi + fg * pm);
    mem_f32[((long)t * kB + b) * kC + n] = mem;

    // a at (row b, col n) -> A-fragment position for step_proj
    const int mt2   = b >> 4;
    const int kt2   = n >> 5;
    const int kk2   = n & 31;
    const int half2 = (kk2 >> 3) & 1;
    const int s2    = ((kk2 >> 4) << 3) + (kk2 & 7);
    const int lane2 = half2 * 16 + (b & 15);
    a_frag[((long)(mt2 * 256 + kt2) * 32 + lane2) * 16 + s2] =
        (__bf16)(og * tanhf(mem));
  }
}

// ---------------------------------------------------------------------------
// Recurrent step, part 2: h = clip(a @ W_proj^T)  [32,4096]x[4096,512]->[32,512]
// Split-K: block = 512 threads (16 waves), one N-tile per block; each wave
// takes 16 K-tiles (32 WMMAs), partials reduced deterministically in LDS.
// grid = 32 blocks x 16 waves = 512 waves.
// ---------------------------------------------------------------------------
__global__ __launch_bounds__(512)
void step_proj(const __bf16* __restrict__ a_frag,  // [2][256][512] A-frags
               const __bf16* __restrict__ pWpj,    // B-frags [32,256]
               float* __restrict__ st_f32,         // [T,B,H]
               float* __restrict__ out,            // [B,T,H]
               int t) {
  __shared__ float s_p[16 * 512];  // 32 KB: per-wave partial sums

  const int tid  = threadIdx.x;
  const int wave = tid >> 5;
  const int lane = tid & 31;
  const int half = lane >> 4;
  const int nt   = blockIdx.x;     // 0..31
  const int k0   = wave * 16;      // this wave's K-chunk (16 of 256 kt)

  const __bf16* a0base = a_frag + ((long)(0 * 256 + k0)) * 512 + (lane << 4);
  const __bf16* a1base = a_frag + ((long)(1 * 256 + k0)) * 512 + (lane << 4);
  const __bf16* bbase  = pWpj + ((long)(nt * 256 + k0)) * 512 + (lane << 4);

  v8f acc0 = {}, acc1 = {};
#pragma unroll 4
  for (int kk = 0; kk < 16; ++kk) {
    if (kk + 1 < 16) __builtin_prefetch(bbase + (kk + 1) * 512, 0, 3);
    const v16bf A0 = *reinterpret_cast<const v16bf*>(a0base + kk * 512);
    const v16bf A1 = *reinterpret_cast<const v16bf*>(a1base + kk * 512);
    const v16bf Bf = *reinterpret_cast<const v16bf*>(bbase  + kk * 512);
    acc0 = wmma_bf16(A0, Bf, acc0);
    acc1 = wmma_bf16(A1, Bf, acc1);
  }
  // Publish partials: s_p[wave*512 + m*16 + n_local]
#pragma unroll
  for (int r = 0; r < 8; ++r) {
    const int m0 = half * 8 + r;
    s_p[wave * 512 + m0 * 16 + (lane & 15)]        = acc0[r];
    s_p[wave * 512 + (m0 + 16) * 16 + (lane & 15)] = acc1[r];
  }
  __syncthreads();

  // Deterministic tree-free reduction: 512 outputs, one per thread.
  const int b  = tid >> 4;   // 0..31
  const int nl = tid & 15;
  float sum = 0.0f;
#pragma unroll
  for (int w = 0; w < 16; ++w) sum += s_p[w * 512 + tid];
  const float h = clip3(sum);
  const int n = nt * 16 + nl;
  st_f32[((long)t * kB + b) * kH + n] = h;
  out[((long)b * kT + t) * kH + n] = h;
}

// ---------------------------------------------------------------------------
extern "C" void kernel_launch(void* const* d_in, const int* in_sizes, int n_in,
                              void* d_out, int out_size, void* d_ws, size_t ws_size,
                              hipStream_t stream) {
  const float* x        = (const float*)d_in[0];  // [B,T,D_IN]
  const int*   prev_idx = (const int*)  d_in[1];  // [B,T,P]
  const float* prev_w   = (const float*)d_in[2];  // [B,T,P]
  const float* W_in     = (const float*)d_in[3];  // [4C, D_IN]
  const float* W_state  = (const float*)d_in[4];  // [4C, H]
  const float* b_state  = (const float*)d_in[5];  // [4C]
  const float* W_proj   = (const float*)d_in[6];  // [H, C]
  float* out = (float*)d_out;                     // [B,T,H]
  (void)in_sizes; (void)n_in; (void)out_size; (void)ws_size;

  // Workspace carve (256B aligned regions). Total ~= 400 MB.
  char* ws = (char*)d_ws;
  size_t off = 0;
  auto carve = [&](size_t bytes) -> void* {
    void* p = ws + off;
    off += (bytes + 255) & ~(size_t)255;
    return p;
  };
  __bf16* pWin  = (__bf16*)carve(sizeof(__bf16) * (size_t)1024 * 16 * 512); // 16.8 MB
  __bf16* pWst  = (__bf16*)carve(sizeof(__bf16) * (size_t)1024 * 16 * 512); // 16.8 MB
  __bf16* pWpj  = (__bf16*)carve(sizeof(__bf16) * (size_t)32 * 256 * 512);  //  8.4 MB
  __bf16* pAx   = (__bf16*)carve(sizeof(__bf16) * (size_t)256 * 16 * 512);  //  4.2 MB
  float*  proj  = (float*) carve(sizeof(float)  * (size_t)kB * kT * kG);    //  256 MB
  float*  memb  = (float*) carve(sizeof(float)  * (size_t)kT * kB * kC);    //   64 MB
  float*  stb   = (float*) carve(sizeof(float)  * (size_t)kT * kB * kH);    //    8 MB
  __bf16* afrag = (__bf16*)carve(sizeof(__bf16) * (size_t)2 * 256 * 512);   //  512 KB

  // One-time operand repack into WMMA fragment order (stays hot in 192MB L2).
  pack_w<<<32768, 256, 0, stream>>>(W_in,    pWin, 1024, 16,  kD);
  pack_w<<<32768, 256, 0, stream>>>(W_state, pWst, 1024, 16,  kH);
  pack_w<<<16384, 256, 0, stream>>>(W_proj,  pWpj, 32,   256, kC);
  pack_a<<<8192,  256, 0, stream>>>(x,       pAx,  256,  16,  kD);

  // Hoisted input projection: [4096,512] x [512,16384].
  gemm_proj_in<<<dim3(128, 128), 256, 0, stream>>>(pAx, pWin, proj);

  // Sequential lattice scan: two WMMA kernels per step, serialized on stream.
  for (int t = 0; t < kT; ++t) {
    step_gate<<<64, 512, 0, stream>>>(proj, prev_idx, prev_w, pWst, b_state,
                                      stb, memb, afrag, t);
    step_proj<<<32, 512, 0, stream>>>(afrag, pWpj, stb, out, t);
  }
}
